// DbrxFFN_14955076125245
// MI455X (gfx1250) — matrix operand order repaired and checked by reading, not compiled, not run on hardware
//
#include <hip/hip_runtime.h>
#include <hip/hip_bf16.h>

// ---------------- problem constants ----------------
#define TOK   4096      // B*S tokens
#define DIM   1024      // D (d_model)
#define NEXP  8         // E experts
#define FF    2048      // F ffn hidden
#define LDSP  40        // LDS row stride in bf16 elements (80B; 16B-aligned chunks)
#define MAXROWS 8704    // 2*TOK + NEXP*64 padding upper bound

typedef __bf16 bf16_t;
typedef __attribute__((ext_vector_type(8)))  __bf16 v8bf;
typedef __attribute__((ext_vector_type(16))) __bf16 v16bf;
typedef __attribute__((ext_vector_type(8)))  float  v8f;

// ---------- gfx1250 async global->LDS copy (ASYNCcnt path), guarded ----------
#if defined(__has_builtin)
#if __has_builtin(__builtin_amdgcn_global_load_async_to_lds_b128)
#define HAVE_ASYNC_LDS 1
#endif
#if __has_builtin(__builtin_amdgcn_s_wait_asynccnt)
#define WAIT_ASYNC() __builtin_amdgcn_s_wait_asynccnt(0)
#endif
#endif
#ifndef WAIT_ASYNC
#if defined(__AMDGCN__)
#define WAIT_ASYNC() asm volatile("s_wait_asynccnt 0" ::: "memory")
#else
#define WAIT_ASYNC()
#endif
#endif

#ifdef HAVE_ASYNC_LDS
// Builtin signature (from clang diagnostic): (v4i as(1)*, v4i as(3)*, imm off, imm cpol)
typedef int v4i __attribute__((vector_size(16)));
typedef __attribute__((address_space(1))) v4i gv4i_t;
typedef __attribute__((address_space(3))) v4i lv4i_t;
__device__ __forceinline__ void async_copy16(const void* g, void* l) {
  // global_load_async_to_lds_b128: per-lane 16B, tracked by ASYNCcnt
  __builtin_amdgcn_global_load_async_to_lds_b128((gv4i_t*)g, (lv4i_t*)l, 0, 0);
}
#endif

__device__ __forceinline__ v8f wmma_bf16f32(v16bf a, v16bf b, v8f c) {
  // v_wmma_f32_16x16x32_bf16: (neg_a, A, neg_b, B, c_mod, C, reuse_a, reuse_b)
  return __builtin_amdgcn_wmma_f32_16x16x32_bf16(false, a, false, b, (short)0, c,
                                                 false, false);
}

// A fragment (16x32 bf16, MxK): lane m = lane&15, half = lane>>4.
// a[0..7]  = K = 8*half + 0..7 ; a[8..15] = K = 16 + 8*half + 0..7
__device__ __forceinline__ v16bf load_a_frag(const bf16_t* As, int row, int half) {
  v16bf a;
  ((v8bf*)&a)[0] = *(const v8bf*)(As + row * LDSP + 8 * half);
  ((v8bf*)&a)[1] = *(const v8bf*)(As + row * LDSP + 16 + 8 * half);
  return a;
}
// B fragment (32x16 bf16, KxN): lane n = lane&15, half = lane>>4.
// b[0..7] = K = 8*half + 0..7 ; b[8..15] = K = 16 + 8*half + 0..7
__device__ __forceinline__ v16bf load_b_frag(const bf16_t* Bs, int n, int half) {
  v16bf b;
  ((v8bf*)&b)[0] = *(const v8bf*)(Bs + n * LDSP + 8 * half);
  ((v8bf*)&b)[1] = *(const v8bf*)(Bs + n * LDSP + 16 + 8 * half);
  return b;
}

__device__ __forceinline__ v8f zero_v8f() {
  v8f z = {0.f, 0.f, 0.f, 0.f, 0.f, 0.f, 0.f, 0.f};
  return z;
}
__device__ __forceinline__ v8bf zero_v8bf() {
  v8bf z = {(__bf16)0.f, (__bf16)0.f, (__bf16)0.f, (__bf16)0.f,
            (__bf16)0.f, (__bf16)0.f, (__bf16)0.f, (__bf16)0.f};
  return z;
}

// ---------------- init: zero output + expert counts ----------------
__global__ void init_kernel(float* __restrict__ out, int n4, int* __restrict__ counts) {
  int i = blockIdx.x * blockDim.x + threadIdx.x;
  if (i < n4) ((float4*)out)[i] = make_float4(0.f, 0.f, 0.f, 0.f);
  if (blockIdx.x == 0 && threadIdx.x < NEXP) counts[threadIdx.x] = 0;
}

// ---------------- fp32 -> bf16 elementwise convert ----------------
__global__ void cvt_bf16_kernel(const float* __restrict__ src, bf16_t* __restrict__ dst,
                                int n) {
  int i = (blockIdx.x * blockDim.x + threadIdx.x) * 8;
  if (i + 8 <= n) {
    float4 f0 = *(const float4*)(src + i);
    float4 f1 = *(const float4*)(src + i + 4);
    v8bf o;
    o[0] = (__bf16)f0.x; o[1] = (__bf16)f0.y; o[2] = (__bf16)f0.z; o[3] = (__bf16)f0.w;
    o[4] = (__bf16)f1.x; o[5] = (__bf16)f1.y; o[6] = (__bf16)f1.z; o[7] = (__bf16)f1.w;
    *(v8bf*)(dst + i) = o;
  }
}

// ---------------- w2 (E*F, D) fp32 -> w2t (E, D, F) bf16, LDS-tiled transpose ----
__global__ __launch_bounds__(256) void w2_transpose_kernel(const float* __restrict__ w2,
                                                           bf16_t* __restrict__ w2t) {
  __shared__ bf16_t tile[32][33];
  const int e = blockIdx.z;
  const int f0 = blockIdx.y * 32;
  const int d0 = blockIdx.x * 32;
  const int tid = threadIdx.x;
  const float* src = w2 + ((size_t)e * FF + f0) * DIM + d0;
#pragma unroll
  for (int k = 0; k < 4; ++k) {
    int lin = tid + k * 256;
    int r = lin >> 5, c = lin & 31;           // r = f-local, c = d-local (coalesced read)
    tile[r][c] = (__bf16)src[(size_t)r * DIM + c];
  }
  __syncthreads();
  bf16_t* dst = w2t + ((size_t)e * DIM + d0) * FF + f0;
#pragma unroll
  for (int k = 0; k < 4; ++k) {
    int lin = tid + k * 256;
    int r = lin >> 5, c = lin & 31;           // r = d-local, c = f-local (coalesced write)
    dst[(size_t)r * FF + c] = tile[c][r];
  }
}

// ---------------- router: logits, softmax, top-2, L1 renorm, counts ----------------
__global__ __launch_bounds__(256) void router_kernel(const float* __restrict__ x,
                                                     const float* __restrict__ rw,
                                                     float* __restrict__ gates,
                                                     int* __restrict__ counts) {
  const int lane = threadIdx.x & 31;
  const int wv = threadIdx.x >> 5;
  const int t = blockIdx.x * 8 + wv;          // one wave32 per token
  float part[NEXP];
#pragma unroll
  for (int e = 0; e < NEXP; ++e) part[e] = 0.f;
  for (int d = lane; d < DIM; d += 32) {
    float xv = x[(size_t)t * DIM + d];
#pragma unroll
    for (int e = 0; e < NEXP; ++e) part[e] += xv * rw[(size_t)e * DIM + d];
  }
#pragma unroll
  for (int e = 0; e < NEXP; ++e) {
#pragma unroll
    for (int off = 16; off > 0; off >>= 1) part[e] += __shfl_xor(part[e], off, 32);
  }
  if (lane == 0) {
    float m = part[0];
#pragma unroll
    for (int e = 1; e < NEXP; ++e) m = fmaxf(m, part[e]);
    float ex[NEXP], s = 0.f;
#pragma unroll
    for (int e = 0; e < NEXP; ++e) { ex[e] = __expf(part[e] - m); s += ex[e]; }
    float w[NEXP];
#pragma unroll
    for (int e = 0; e < NEXP; ++e) w[e] = ex[e] / s;
    int i0 = 0;
#pragma unroll
    for (int e = 1; e < NEXP; ++e) if (w[e] > w[i0]) i0 = e;
    int i1 = (i0 == 0) ? 1 : 0;
#pragma unroll
    for (int e = 0; e < NEXP; ++e) if (e != i0 && w[e] > w[i1]) i1 = e;
    float s2 = w[i0] + w[i1];
#pragma unroll
    for (int e = 0; e < NEXP; ++e) gates[(size_t)t * NEXP + e] = 0.f;
    gates[(size_t)t * NEXP + i0] = w[i0] / s2;
    gates[(size_t)t * NEXP + i1] = w[i1] / s2;
    atomicAdd(&counts[i0], 1);
    atomicAdd(&counts[i1], 1);
  }
}

// ---------------- padded exclusive prefix over counts ----------------
__global__ void offsets_kernel(const int* __restrict__ counts, int* __restrict__ offs,
                               int* __restrict__ cursor) {
  if (blockIdx.x == 0 && threadIdx.x == 0) {
    int o = 0;
    for (int e = 0; e < NEXP; ++e) {
      offs[e] = o;
      cursor[e] = o;
      o += ((counts[e] + 63) >> 6) << 6;      // pad each expert to 64-row tiles
    }
  }
}

// ---------------- scatter token ids into per-expert lists ----------------
__global__ void scatter_kernel(const float* __restrict__ gates, int* __restrict__ cursor,
                               int* __restrict__ tok_ids) {
  int t = blockIdx.x * blockDim.x + threadIdx.x;
  if (t < TOK) {
#pragma unroll
    for (int e = 0; e < NEXP; ++e) {
      if (gates[(size_t)t * NEXP + e] > 0.f) {
        int p = atomicAdd(&cursor[e], 1);
        tok_ids[p] = t;
      }
    }
  }
}

// ---------------- h = silu(x@w1^T) * (x@v1^T), gathered, bf16, fused dual-GEMM ----
// grid: (FF/128, 64, NEXP), block 256 (8 wave32: 2(M) x 4(N)), tile 64M x 128N x 32K
// Double-buffered LDS; async global->LDS staging; one barrier per K-step.
__global__ __launch_bounds__(256) void moe_h_kernel(
    const bf16_t* __restrict__ xb, const bf16_t* __restrict__ w1b,
    const bf16_t* __restrict__ v1b, bf16_t* __restrict__ hbuf,
    const int* __restrict__ counts, const int* __restrict__ offs,
    const int* __restrict__ tok_ids) {
  const int e = blockIdx.z;
  const int cnt = counts[e];
  const int m0 = blockIdx.y * 64;
  if (m0 >= cnt) return;
  const int off_e = offs[e];
  const int f0 = blockIdx.x * 128;
  const int tid = threadIdx.x;
  const int lane = tid & 31;
  const int wv = tid >> 5;
  const int wm = wv & 1, wn = wv >> 1;
  const int valid = cnt - m0;

  __shared__ bf16_t AsB[2][64 * LDSP];
  __shared__ bf16_t BwB[2][128 * LDSP];
  __shared__ bf16_t BvB[2][128 * LDSP];

  const bf16_t* w1e = w1b + (size_t)e * FF * DIM;
  const bf16_t* v1e = v1b + (size_t)e * FF * DIM;

  // A staging assignment: thread -> (row, 16B chunk); token index fixed per thread
  const int ar = tid >> 2, ac = (tid & 3) * 8;
  const bf16_t* asrc = nullptr;
  if (ar < valid) asrc = xb + (size_t)tok_ids[off_e + m0 + ar] * DIM + ac;

  auto stage = [&](int buf, int k0) {
    bf16_t* As = &AsB[buf][0];
    bf16_t* Bw = &BwB[buf][0];
    bf16_t* Bv = &BvB[buf][0];
#ifdef HAVE_ASYNC_LDS
    if (asrc)
      async_copy16(asrc + k0, As + ar * LDSP + ac);
    else
      *(v8bf*)(As + ar * LDSP + ac) = zero_v8bf();
#pragma unroll
    for (int s = 0; s < 2; ++s) {
      int lin = tid + s * 256;
      int br = lin >> 2, bc = (lin & 3) * 8;
      async_copy16(w1e + (size_t)(f0 + br) * DIM + k0 + bc, Bw + br * LDSP + bc);
      async_copy16(v1e + (size_t)(f0 + br) * DIM + k0 + bc, Bv + br * LDSP + bc);
    }
#else
    v8bf av = zero_v8bf();
    if (asrc) av = *(const v8bf*)(asrc + k0);
    *(v8bf*)(As + ar * LDSP + ac) = av;
#pragma unroll
    for (int s = 0; s < 2; ++s) {
      int lin = tid + s * 256;
      int br = lin >> 2, bc = (lin & 3) * 8;
      *(v8bf*)(Bw + br * LDSP + bc) =
          *(const v8bf*)(w1e + (size_t)(f0 + br) * DIM + k0 + bc);
      *(v8bf*)(Bv + br * LDSP + bc) =
          *(const v8bf*)(v1e + (size_t)(f0 + br) * DIM + k0 + bc);
    }
#endif
  };

  v8f accw[2][2], accv[2][2];
#pragma unroll
  for (int i = 0; i < 2; ++i)
#pragma unroll
    for (int j = 0; j < 2; ++j) { accw[i][j] = zero_v8f(); accv[i][j] = zero_v8f(); }

  const int half = lane >> 4, nl = lane & 15, ml = lane & 15;
  const int NS = DIM / 32;

  stage(0, 0);
  for (int ks = 0; ks < NS; ++ks) {
    WAIT_ASYNC();          // own async stage of tile `cur` complete
    __syncthreads();       // all waves done staging `cur` & done reading `nxt`
    if (ks + 1 < NS) stage((ks + 1) & 1, (ks + 1) * 32);   // prefetch next tile
    const int cur = ks & 1;
    const bf16_t* As = &AsB[cur][0];
    const bf16_t* Bw = &BwB[cur][0];
    const bf16_t* Bv = &BvB[cur][0];
    // load all fragments, then a dense WMMA burst
    v16bf a0 = load_a_frag(As, wm * 32 + ml, half);
    v16bf a1 = load_a_frag(As, wm * 32 + 16 + ml, half);
    v16bf bw0 = load_b_frag(Bw, wn * 32 + nl, half);
    v16bf bw1 = load_b_frag(Bw, wn * 32 + 16 + nl, half);
    v16bf bv0 = load_b_frag(Bv, wn * 32 + nl, half);
    v16bf bv1 = load_b_frag(Bv, wn * 32 + 16 + nl, half);
    accw[0][0] = wmma_bf16f32(a0, bw0, accw[0][0]);
    accw[1][0] = wmma_bf16f32(a1, bw0, accw[1][0]);
    accw[0][1] = wmma_bf16f32(a0, bw1, accw[0][1]);
    accw[1][1] = wmma_bf16f32(a1, bw1, accw[1][1]);
    accv[0][0] = wmma_bf16f32(a0, bv0, accv[0][0]);
    accv[1][0] = wmma_bf16f32(a1, bv0, accv[1][0]);
    accv[0][1] = wmma_bf16f32(a0, bv1, accv[0][1]);
    accv[1][1] = wmma_bf16f32(a1, bv1, accv[1][1]);
  }

  // C/D layout: vgpr i -> M = i + 8*half, N = lane&15
#pragma unroll
  for (int mt = 0; mt < 2; ++mt)
#pragma unroll
    for (int nt = 0; nt < 2; ++nt) {
      int grow = off_e + m0 + wm * 32 + mt * 16 + 8 * half;
      int gcol = f0 + wn * 32 + nt * 16 + nl;
#pragma unroll
      for (int i = 0; i < 8; ++i) {
        float aw = accw[mt][nt][i];
        float sl = aw / (1.f + __expf(-aw));   // silu
        float hv = sl * accv[mt][nt][i];
        hbuf[(size_t)(grow + i) * FF + gcol] = (bf16_t)hv;
      }
    }
}

// ---------------- out[t] += gate * (h @ w2t_e^T), one expert per launch ----------
// grid: (DIM/128, 64), block 256, tile 64M x 128N x 32K over K=FF
__global__ __launch_bounds__(256) void moe_y_kernel(
    const bf16_t* __restrict__ hbuf, const bf16_t* __restrict__ w2tb,
    const float* __restrict__ gates, const int* __restrict__ counts,
    const int* __restrict__ offs, const int* __restrict__ tok_ids,
    float* __restrict__ out, int e) {
  const int cnt = counts[e];
  const int m0 = blockIdx.y * 64;
  if (m0 >= cnt) return;
  const int off_e = offs[e];
  const int d0 = blockIdx.x * 128;
  const int tid = threadIdx.x;
  const int lane = tid & 31;
  const int wv = tid >> 5;
  const int wm = wv & 1, wn = wv >> 1;
  const int valid = cnt - m0;

  __shared__ bf16_t AsB[2][64 * LDSP];
  __shared__ bf16_t BsB[2][128 * LDSP];

  const bf16_t* w2e = w2tb + (size_t)e * DIM * FF;    // (D, F) row-major
  const bf16_t* hA = hbuf + (size_t)(off_e + m0) * FF;

  const int ar = tid >> 2, ac = (tid & 3) * 8;

  auto stage = [&](int buf, int k0) {
    bf16_t* As = &AsB[buf][0];
    bf16_t* Bs = &BsB[buf][0];
#ifdef HAVE_ASYNC_LDS
    async_copy16(hA + (size_t)ar * FF + k0 + ac, As + ar * LDSP + ac);
#pragma unroll
    for (int s = 0; s < 2; ++s) {
      int lin = tid + s * 256;
      int br = lin >> 2, bc = (lin & 3) * 8;
      async_copy16(w2e + (size_t)(d0 + br) * FF + k0 + bc, Bs + br * LDSP + bc);
    }
#else
    *(v8bf*)(As + ar * LDSP + ac) = *(const v8bf*)(hA + (size_t)ar * FF + k0 + ac);
#pragma unroll
    for (int s = 0; s < 2; ++s) {
      int lin = tid + s * 256;
      int br = lin >> 2, bc = (lin & 3) * 8;
      *(v8bf*)(Bs + br * LDSP + bc) =
          *(const v8bf*)(w2e + (size_t)(d0 + br) * FF + k0 + bc);
    }
#endif
  };

  v8f acc[2][2];
#pragma unroll
  for (int i = 0; i < 2; ++i)
#pragma unroll
    for (int j = 0; j < 2; ++j) acc[i][j] = zero_v8f();

  const int half = lane >> 4, nl = lane & 15, ml = lane & 15;
  const int NS = FF / 32;

  stage(0, 0);
  for (int ks = 0; ks < NS; ++ks) {
    WAIT_ASYNC();
    __syncthreads();
    if (ks + 1 < NS) stage((ks + 1) & 1, (ks + 1) * 32);
    const int cur = ks & 1;
    const bf16_t* As = &AsB[cur][0];
    const bf16_t* Bs = &BsB[cur][0];
    v16bf a0 = load_a_frag(As, wm * 32 + ml, half);
    v16bf a1 = load_a_frag(As, wm * 32 + 16 + ml, half);
    v16bf b0 = load_b_frag(Bs, wn * 32 + nl, half);
    v16bf b1 = load_b_frag(Bs, wn * 32 + 16 + nl, half);
    acc[0][0] = wmma_bf16f32(a0, b0, acc[0][0]);
    acc[1][0] = wmma_bf16f32(a1, b0, acc[1][0]);
    acc[0][1] = wmma_bf16f32(a0, b1, acc[0][1]);
    acc[1][1] = wmma_bf16f32(a1, b1, acc[1][1]);
  }

#pragma unroll
  for (int mt = 0; mt < 2; ++mt)
#pragma unroll
    for (int nt = 0; nt < 2; ++nt) {
      int lrow0 = wm * 32 + mt * 16 + 8 * half;
      int gcol = d0 + wn * 32 + nt * 16 + nl;
#pragma unroll
      for (int i = 0; i < 8; ++i) {
        int lr = lrow0 + i;
        if (lr < valid) {
          int t = tok_ids[off_e + m0 + lr];
          float g = gates[(size_t)t * NEXP + e];
          out[(size_t)t * DIM + gcol] += g * acc[mt][nt][i];   // race-free: per-expert launch
        }
      }
    }
}

// ---------------- host launcher ----------------
extern "C" void kernel_launch(void* const* d_in, const int* in_sizes, int n_in,
                              void* d_out, int out_size, void* d_ws, size_t ws_size,
                              hipStream_t stream) {
  (void)in_sizes; (void)n_in; (void)out_size; (void)ws_size;
  const float* x  = (const float*)d_in[0];
  const float* w1 = (const float*)d_in[1];
  const float* v1 = (const float*)d_in[2];
  const float* w2 = (const float*)d_in[3];
  const float* rw = (const float*)d_in[4];
  float* out = (float*)d_out;

  char* ws = (char*)d_ws;
  size_t off = 0;
  auto take = [&](size_t bytes) -> char* {
    off = (off + 255) & ~(size_t)255;
    char* p = ws + off;
    off += bytes;
    return p;
  };
  const size_t nXD = (size_t)TOK * DIM;          // 4,194,304
  const size_t nEFD = (size_t)NEXP * FF * DIM;   // 16,777,216

  bf16_t* xb     = (bf16_t*)take(nXD * 2);
  bf16_t* w1b    = (bf16_t*)take(nEFD * 2);
  bf16_t* v1b    = (bf16_t*)take(nEFD * 2);
  bf16_t* w2tb   = (bf16_t*)take(nEFD * 2);
  float*  gates  = (float*)take((size_t)TOK * NEXP * 4);
  int*    counts = (int*)take(NEXP * 4);
  int*    offs   = (int*)take(NEXP * 4);
  int*    cursor = (int*)take(NEXP * 4);
  int*    tokids = (int*)take((size_t)MAXROWS * 4);
  bf16_t* hbuf   = (bf16_t*)take((size_t)MAXROWS * FF * 2);

  // 1. zero output + counts
  init_kernel<<<(int)(nXD / 4 / 256), 256, 0, stream>>>(out, (int)(nXD / 4), counts);
  // 2. bf16 conversions (weights become L2-resident bf16)
  cvt_bf16_kernel<<<(int)(nXD / 8 / 256), 256, 0, stream>>>(x, xb, (int)nXD);
  cvt_bf16_kernel<<<(int)(nEFD / 8 / 256), 256, 0, stream>>>(w1, w1b, (int)nEFD);
  cvt_bf16_kernel<<<(int)(nEFD / 8 / 256), 256, 0, stream>>>(v1, v1b, (int)nEFD);
  w2_transpose_kernel<<<dim3(DIM / 32, FF / 32, NEXP), 256, 0, stream>>>(w2, w2tb);
  // 3. routing: gates + per-expert token lists
  router_kernel<<<TOK / 8, 256, 0, stream>>>(x, rw, gates, counts);
  offsets_kernel<<<1, 32, 0, stream>>>(counts, offs, cursor);
  scatter_kernel<<<TOK / 256, 256, 0, stream>>>(gates, cursor, tokids);
  // 4. gathered dual-GEMM + SwiGLU (all experts in one launch; disjoint writes)
  moe_h_kernel<<<dim3(FF / 128, 64, NEXP), 256, 0, stream>>>(xb, w1b, v1b, hbuf,
                                                             counts, offs, tokids);
  // 5. gathered down-projection, gated accumulate (per-expert launches => deterministic)
  for (int e = 0; e < NEXP; ++e)
    moe_y_kernel<<<dim3(DIM / 128, 64), 256, 0, stream>>>(hbuf, w2tb, gates, counts,
                                                          offs, tokids, out, e);
}
